// Model_43224550867783
// MI455X (gfx1250) — matrix operand order repaired and checked by reading, not compiled, not run on hardware
//
#include <hip/hip_runtime.h>
#include <hip/hip_bf16.h>
#include <math.h>

// ---------------------------------------------------------------------------
// CDNA5 (gfx1250) wave32 WMMA types
// ---------------------------------------------------------------------------
typedef __attribute__((ext_vector_type(16))) __bf16        v16bf;
typedef __attribute__((ext_vector_type(2)))  __bf16        v2bf;
typedef __attribute__((ext_vector_type(8)))  float         v8f;
typedef __attribute__((ext_vector_type(8)))  unsigned int  v8u;

#define WMMA_BF16(a, b, c) \
  __builtin_amdgcn_wmma_f32_16x16x32_bf16(false, (a), false, (b), (short)0, (c), false, false)

__device__ __forceinline__ unsigned short bf16_bits(float x) {
  __bf16 h = (__bf16)x;
  return __builtin_bit_cast(unsigned short, h);
}
// Pack two floats into one u32 of 2xbf16 — shaped so the backend matches a
// single v_cvt_pk_bf16_f32 (build_vector of two fptrunc, then bitcast).
__device__ __forceinline__ unsigned int pack2_bf16(float x, float y) {
  v2bf v;
  v[0] = (__bf16)x;
  v[1] = (__bf16)y;
  return __builtin_bit_cast(unsigned int, v);
}

// ---------------------------------------------------------------------------
// Fast (unguarded) fragment loaders: 4 x float4 loads each, no branches.
// Preconditions: rows [row0,row0+16) x cols [k0,k0+32) readable; ld % 4 == 0.
//
// A (16x32): lane m = lane&15, half = lane>>4. Element pairs sit at row
// offsets {0,2,4,6} and {16,18,20,22} + half*8 -> two runs of 8 consecutive
// floats: [base..base+7] -> f[0..7], [base+16..base+23] -> f[8..15].
// ---------------------------------------------------------------------------
__device__ __forceinline__ v16bf load_a_frag_fast(const float* __restrict__ A, int lda,
                                                  int row0, int k0, int lane) {
  int m = row0 + (lane & 15);
  int half = lane >> 4;
  const float* p = A + (size_t)m * lda + k0 + half * 8;
  float4 q0 = *(const float4*)(p);
  float4 q1 = *(const float4*)(p + 4);
  float4 q2 = *(const float4*)(p + 16);
  float4 q3 = *(const float4*)(p + 20);
  v8u u;
  u[0] = pack2_bf16(q0.x, q0.y); u[1] = pack2_bf16(q0.z, q0.w);
  u[2] = pack2_bf16(q1.x, q1.y); u[3] = pack2_bf16(q1.z, q1.w);
  u[4] = pack2_bf16(q2.x, q2.y); u[5] = pack2_bf16(q2.z, q2.w);
  u[6] = pack2_bf16(q3.x, q3.y); u[7] = pack2_bf16(q3.z, q3.w);
  return __builtin_bit_cast(v16bf, u);
}

// B for D = A*B with Bfrag[k][n] = W[n][k]: lane n = lane&15, half = lane>>4;
// f[0..7] = W[n0+n][k0 + half*8 .. +7], f[8..15] = W[n0+n][k0 + 16 + half*8 .. +7].
__device__ __forceinline__ v16bf load_b_frag_fast(const float* __restrict__ W, int ldw,
                                                  int n0, int k0, int lane) {
  int n = n0 + (lane & 15);
  int half = lane >> 4;
  const float* p = W + (size_t)n * ldw + k0 + half * 8;
  float4 q0 = *(const float4*)(p);
  float4 q1 = *(const float4*)(p + 4);
  float4 q2 = *(const float4*)(p + 16);
  float4 q3 = *(const float4*)(p + 20);
  v8u u;
  u[0] = pack2_bf16(q0.x, q0.y); u[1] = pack2_bf16(q0.z, q0.w);
  u[2] = pack2_bf16(q1.x, q1.y); u[3] = pack2_bf16(q1.z, q1.w);
  u[4] = pack2_bf16(q2.x, q2.y); u[5] = pack2_bf16(q2.z, q2.w);
  u[6] = pack2_bf16(q3.x, q3.y); u[7] = pack2_bf16(q3.z, q3.w);
  return __builtin_bit_cast(v16bf, u);
}

// ---------------------------------------------------------------------------
// Generic GEMM: C[M,N] = act(A[M,K] @ W[N,K]^T + bias), f32 in/out, bf16 WMMA.
// Block = 256 threads = 8 waves (2 M-tiles x 4 N-tiles); block tile 32x64.
// Requires M%32==0, N%64==0, lda/ldw %4==0 (true for every call site).
// Main K loop is branchless: 3 float4 global loads + uint2/uint4 LDS stores
// per thread; one guarded tail chunk handles K%32 != 0 (conv1 only).
// Fragment-order LDS: each thread's staged values are contiguous half-words;
// each lane reads its whole 16-element fragment as 32B of LDS.
// ---------------------------------------------------------------------------
__global__ void gemm_bias_act(const float* __restrict__ A, const float* __restrict__ W,
                              const float* __restrict__ bias, float* __restrict__ C,
                              int M, int N, int K, int lda, int ldw, int ldc, int act) {
  __shared__ __align__(16) unsigned short aL[2][32][16];  // [wm][lane][fi]
  __shared__ __align__(16) unsigned short bL[4][32][16];  // [wn][lane][fi]
  int tid = threadIdx.x;
  int lane = tid & 31;
  int wv = tid >> 5;
  int wm = wv >> 2, wn = wv & 3;
  int blockM = blockIdx.y * 32;
  int blockN = blockIdx.x * 64;

  // ---- A staging coords: row aml, 4 consecutive k at akq ----
  int aml = tid >> 3;              // 0..31
  int akq = (tid & 7) * 4;         // 0,4,...,28
  int akp = akq >> 1;              // even
  int a_half = (akp >> 2) & 1;
  int a_v = (akp & 3) | (((akp >> 3) & 1) << 2);   // even
  unsigned int* aDst = (unsigned int*)&aL[aml >> 4][(aml & 15) + 16 * a_half][2 * a_v];
  const float* aPtr = A + (size_t)(blockM + aml) * lda + akq;

  // ---- B staging coords: row bnl, 8 consecutive k at bkq ----
  int bnl = tid >> 2;              // 0..63
  int bkq = (tid & 3) * 8;         // 0,8,16,24
  unsigned int* bDst = (unsigned int*)&bL[bnl >> 4][(bnl & 15) + 16 * ((bkq >> 3) & 1)][(bkq >> 4) * 8];
  const float* bPtr = W + (size_t)(blockN + bnl) * ldw + bkq;

  v8f acc = {};
  int Kmain = K & ~31;
  for (int k0 = 0; k0 < Kmain; k0 += 32) {
    float4 av  = *(const float4*)aPtr;        aPtr += 32;
    float4 bv0 = *(const float4*)bPtr;
    float4 bv1 = *(const float4*)(bPtr + 4);  bPtr += 32;
    uint2 ap; ap.x = pack2_bf16(av.x, av.y); ap.y = pack2_bf16(av.z, av.w);
    *(uint2*)aDst = ap;
    uint4 bp;
    bp.x = pack2_bf16(bv0.x, bv0.y); bp.y = pack2_bf16(bv0.z, bv0.w);
    bp.z = pack2_bf16(bv1.x, bv1.y); bp.w = pack2_bf16(bv1.z, bv1.w);
    *(uint4*)bDst = bp;
    __syncthreads();
    v16bf a = *reinterpret_cast<const v16bf*>(&aL[wm][lane][0]);
    v16bf b = *reinterpret_cast<const v16bf*>(&bL[wn][lane][0]);
    acc = WMMA_BF16(a, b, acc);
    __syncthreads();
  }
  if (Kmain < K) {  // guarded tail chunk (conv1: K=240)
    unsigned short* aDstH = (unsigned short*)aDst;
    unsigned short* bDstH = (unsigned short*)bDst;
#pragma unroll
    for (int j = 0; j < 4; ++j) {
      int kg = Kmain + akq + j;
      float v = (kg < K) ? aPtr[j] : 0.f;   // aPtr already at Kmain+akq
      aDstH[j] = bf16_bits(v);
    }
#pragma unroll
    for (int j = 0; j < 8; ++j) {
      int kg = Kmain + bkq + j;
      float v = (kg < K) ? bPtr[j] : 0.f;
      bDstH[j] = bf16_bits(v);
    }
    __syncthreads();
    v16bf a = *reinterpret_cast<const v16bf*>(&aL[wm][lane][0]);
    v16bf b = *reinterpret_cast<const v16bf*>(&bL[wn][lane][0]);
    acc = WMMA_BF16(a, b, acc);
    __syncthreads();
  }

  int half = lane >> 4;
  int n = blockN + wn * 16 + (lane & 15);
  float bs = bias ? bias[n] : 0.f;
#pragma unroll
  for (int v = 0; v < 8; ++v) {
    int m = blockM + wm * 16 + v + 8 * half;
    float val = acc[v] + bs;
    if (act) val = fmaxf(val, 0.f);
    C[(size_t)m * ldc + n] = val;
  }
}

// ---------------------------------------------------------------------------
// Fused GRU step: gh = h_prev @ Whh^T + bhh (WMMA), then gate update.
// One wave per 16 hidden units. h buffers are padded to 16 rows (rows 8..15
// zero) so the A fragment loads are unguarded float4 loads.
// ---------------------------------------------------------------------------
__global__ void gru_step(const float* __restrict__ xp_t, long xp_bstride,
                         const float* __restrict__ Whh, const float* __restrict__ bhh,
                         const float* __restrict__ h_prev, float* __restrict__ h_next,
                         float* __restrict__ hs_t, long hs_bstride, int H) {
  int lane = threadIdx.x & 31;
  int jt = blockIdx.x * 16;
  v8f ar = {}, az = {}, an_ = {};
  for (int k0 = 0; k0 < H; k0 += 32) {
    v16bf a  = load_a_frag_fast(h_prev, H, 0, k0, lane);
    v16bf br = load_b_frag_fast(Whh, H, jt,         k0, lane);
    v16bf bz = load_b_frag_fast(Whh, H, H + jt,     k0, lane);
    v16bf bn = load_b_frag_fast(Whh, H, 2 * H + jt, k0, lane);
    ar  = WMMA_BF16(a, br, ar);
    az  = WMMA_BF16(a, bz, az);
    an_ = WMMA_BF16(a, bn, an_);
  }
  // D layout: lanes 0-15, VGPR v -> (m=v, n=lane): lane<16 owns hidden unit
  // j = jt+lane with all 8 batch rows in acc[0..7].
  if (lane < 16) {
    int j = jt + lane;
    float bR = bhh[j], bZ = bhh[H + j], bN = bhh[2 * H + j];
#pragma unroll
    for (int b = 0; b < 8; ++b) {
      float hr = ar[b] + bR;
      float hz = az[b] + bZ;
      float hn = an_[b] + bN;
      float xr = xp_t[(long)b * xp_bstride + j];
      float xz = xp_t[(long)b * xp_bstride + H + j];
      float xn = xp_t[(long)b * xp_bstride + 2 * H + j];
      float r  = 1.f / (1.f + __expf(-(xr + hr)));
      float z  = 1.f / (1.f + __expf(-(xz + hz)));
      float nn = tanhf(xn + r * hn);
      float hp = h_prev[(long)b * H + j];
      float hv = (1.f - z) * nn + z * hp;
      h_next[(long)b * H + j] = hv;
      hs_t[(long)b * hs_bstride + j] = hv;
    }
  }
}

// ---------------------------------------------------------------------------
// Scalar helper kernels
// ---------------------------------------------------------------------------
__global__ void fill_f32(float* p, long n, float v) {
  long i = (long)blockIdx.x * blockDim.x + threadIdx.x;
  if (i < n) p[i] = v;
}

// im2col with arbitrary input element strides (sb, sc, st). Output row-major
// col[(b*Tout+to)][ci*ksz+kk].
__global__ void im2col_k(const float* __restrict__ X, float* __restrict__ col,
                         int Bn, int Cin, int Tin, int Tout, int ksz, int stride, int pad,
                         long sb, long sc, long st) {
  long total = (long)Bn * Tout * Cin * ksz;
  long i = (long)blockIdx.x * blockDim.x + threadIdx.x;
  if (i >= total) return;
  int kk = (int)(i % ksz); long r = i / ksz;
  int ci = (int)(r % Cin); r /= Cin;
  int to = (int)(r % Tout); int b = (int)(r / Tout);
  int ti = to * stride - pad + kk;
  float v = 0.f;
  if (ti >= 0 && ti < Tin) v = X[(long)b * sb + (long)ci * sc + (long)ti * st];
  col[i] = v;  // i == ((b*Tout+to)*Cin+ci)*ksz+kk, matching layout
}

__global__ void bn_stats(const float* __restrict__ Y, float* mean, float* var, int M, int C) {
  int c = blockIdx.x;
  __shared__ float ss[256], sq[256];
  float s = 0.f, q = 0.f;
  for (int m = threadIdx.x; m < M; m += blockDim.x) {
    float v = Y[(size_t)m * C + c];
    s += v; q += v * v;
  }
  ss[threadIdx.x] = s; sq[threadIdx.x] = q; __syncthreads();
  for (int o = blockDim.x >> 1; o > 0; o >>= 1) {
    if (threadIdx.x < (unsigned)o) { ss[threadIdx.x] += ss[threadIdx.x + o]; sq[threadIdx.x] += sq[threadIdx.x + o]; }
    __syncthreads();
  }
  if (threadIdx.x == 0) {
    float mu = ss[0] / M;
    mean[c] = mu;
    var[c] = sq[0] / M - mu * mu;
  }
}

__global__ void bn_relu_k(float* Y, const float* mean, const float* var,
                          const float* g, const float* b, long M, int C) {
  long i = (long)blockIdx.x * blockDim.x + threadIdx.x;
  if (i >= M * C) return;
  int c = (int)(i % C);
  float v = (Y[i] - mean[c]) * rsqrtf(var[c] + 1e-5f) * g[c] + b[c];
  Y[i] = fmaxf(v, 0.f);
}

// VQ: per row argmin over 512 codes (D=64).  ||z||^2 constant dropped.
__global__ void vq_argmin(const float* __restrict__ z, const float* __restrict__ e,
                          int* idx, int* counts) {
  int row = blockIdx.x; int tid = threadIdx.x;  // blockDim=128
  const float* zr = z + (size_t)row * 64;
  float best = 3.4e38f; int bi = 0x7fffffff;
  for (int m = tid; m < 512; m += 128) {
    const float* em = e + (size_t)m * 64;
    float d = 0.f;
    for (int k = 0; k < 64; ++k) { float ev = em[k]; d += ev * ev - 2.f * zr[k] * ev; }
    if (d < best || (d == best && m < bi)) { best = d; bi = m; }
  }
  __shared__ float bv[128]; __shared__ int bx[128];
  bv[tid] = best; bx[tid] = bi; __syncthreads();
  for (int o = 64; o > 0; o >>= 1) {
    if (tid < o) {
      if (bv[tid + o] < bv[tid] || (bv[tid + o] == bv[tid] && bx[tid + o] < bx[tid])) {
        bv[tid] = bv[tid + o]; bx[tid] = bx[tid + o];
      }
    }
    __syncthreads();
  }
  if (tid == 0) { idx[row] = bx[0]; atomicAdd(&counts[bx[0]], 1); }
}

__global__ void vq_gather_loss(const float* __restrict__ z, const float* __restrict__ e,
                               const int* __restrict__ idx, float* q, float* lossAcc) {
  int row = blockIdx.x; int k = threadIdx.x;  // blockDim=64
  float qv = e[(size_t)idx[row] * 64 + k];
  q[(size_t)row * 64 + k] = qv;
  float d = z[(size_t)row * 64 + k] - qv;
  __shared__ float s[64];
  s[k] = d * d; __syncthreads();
  for (int o = 32; o > 0; o >>= 1) { if (k < o) s[k] += s[k + o]; __syncthreads(); }
  if (k == 0) atomicAdd(lossAcc, s[0]);
}

__global__ void vq_finalize(const int* __restrict__ counts, const float* __restrict__ lossAcc,
                            float* out2) {
  __shared__ float s[512];
  int t = threadIdx.x;  // blockDim=512
  float avg = (float)counts[t] / 128.0f;
  s[t] = avg * logf(avg + 1e-10f);
  __syncthreads();
  for (int o = 256; o > 0; o >>= 1) { if (t < o) s[t] += s[t + o]; __syncthreads(); }
  if (t == 0) {
    out2[0] = 0.25f * lossAcc[0] / 8192.0f;  // mean over 128*64 elements
    out2[1] = expf(-s[0]);
  }
}

// h0[b,s,0:64] = q[b, s/2, :]; h0[b,s,64:128] = spk_emb[speakers[b], :]
__global__ void build_h0(const float* __restrict__ q, const float* __restrict__ spk_emb,
                         const int* __restrict__ speakers, float* h0) {
  int i = blockIdx.x * blockDim.x + threadIdx.x;
  if (i >= 8 * 32 * 128) return;
  int c = i & 127; int s = (i >> 7) & 31; int b = i >> 12;
  float v;
  if (c < 64) v = q[((size_t)(b * 16 + (s >> 1))) * 64 + c];
  else        v = spk_emb[(size_t)speakers[b] * 64 + (c - 64)];
  h0[i] = v;
}

__global__ void concat_h(const float* __restrict__ hsF, const float* __restrict__ hsB,
                         float* out, long rows, int H) {
  long i = (long)blockIdx.x * blockDim.x + threadIdx.x;
  if (i >= rows * 2 * H) return;
  int c = (int)(i % (2 * H)); long r = i / (2 * H);
  out[i] = (c < H) ? hsF[r * H + c] : hsB[r * H + (c - H)];
}

// xcat[b,ta,0:256] = aud_emb[x[b,ta]]; xcat[b,ta,256:512] = hcat2[b, 11+ta/200, :]
__global__ void build_xcat(const int* __restrict__ x, const float* __restrict__ aud_emb,
                           const float* __restrict__ hcat2, float* xcat) {
  long i = (long)blockIdx.x * blockDim.x + threadIdx.x;
  if (i >= (long)16000 * 512) return;
  int c = (int)(i & 511); long r = i >> 9;
  int ta = (int)(r % 2000); int b = (int)(r / 2000);
  float v;
  if (c < 256) v = aud_emb[(size_t)x[b * 2000 + ta] * 256 + c];
  else {
    int s = 11 + ta / 200;
    v = hcat2[((size_t)(b * 32 + s)) * 256 + (c - 256)];
  }
  xcat[i] = v;
}

// ---------------------------------------------------------------------------
// Host orchestration
// ---------------------------------------------------------------------------
extern "C" void kernel_launch(void* const* d_in, const int* in_sizes, int n_in,
                              void* d_out, int out_size, void* d_ws, size_t ws_size,
                              hipStream_t stream) {
  // ---- inputs ----
  const int*   x        = (const int*)  d_in[0];   // (8,2000)
  const float* mels     = (const float*)d_in[1];   // (8,80,32)
  const int*   speakers = (const int*)  d_in[2];   // (8,)
  const float* conv_ws[5] = {(const float*)d_in[3], (const float*)d_in[4], (const float*)d_in[5],
                             (const float*)d_in[6], (const float*)d_in[7]};
  const float* bn_g[5] = {(const float*)d_in[8],  (const float*)d_in[9],  (const float*)d_in[10],
                          (const float*)d_in[11], (const float*)d_in[12]};
  const float* bn_b[5] = {(const float*)d_in[13], (const float*)d_in[14], (const float*)d_in[15],
                          (const float*)d_in[16], (const float*)d_in[17]};
  const float* conv6_w  = (const float*)d_in[18];  // (64,512,1)
  const float* conv6_b  = (const float*)d_in[19];
  const float* codebook = (const float*)d_in[20];  // (512,64)
  const float* spk_emb  = (const float*)d_in[21];  // (112,64)
  const float* aud_emb  = (const float*)d_in[22];  // (256,256)
  // rnn1: [l0.fwd(4), l0.bwd(4), l1.fwd(4), l1.bwd(4)] starting at 23
  const float* r1[4][4];
  for (int g = 0; g < 4; ++g)
    for (int k = 0; k < 4; ++k) r1[g][k] = (const float*)d_in[23 + g * 4 + k];
  const float* Wih2 = (const float*)d_in[39];  // (2688,512)
  const float* Whh2 = (const float*)d_in[40];  // (2688,896)
  const float* bih2 = (const float*)d_in[41];
  const float* bhh2 = (const float*)d_in[42];
  const float* fc1_w = (const float*)d_in[43]; // (256,896)
  const float* fc1_b = (const float*)d_in[44];
  const float* fc2_w = (const float*)d_in[45]; // (256,256)
  const float* fc2_b = (const float*)d_in[46];

  float* out  = (float*)d_out;                 // (8,2000,256) flat
  float* out2 = out + (size_t)8 * 2000 * 256;  // [loss, perp]

  // ---- workspace bump allocator ----
  char* wsp = (char*)d_ws;
  auto alloc = [&](size_t nbytes) -> void* {
    void* p = (void*)wsp;
    wsp += (nbytes + 255) & ~(size_t)255;
    return p;
  };
  float* actA   = (float*)alloc(131072 * 4);            // conv act ping (M x 512)
  float* actB   = (float*)alloc(131072 * 4);            // conv act pong
  float* colbuf = (float*)alloc(524288 * 4);            // im2col (max 256x2048)
  float* meanb  = (float*)alloc(512 * 4);
  float* varb   = (float*)alloc(512 * 4);
  float* zbuf   = (float*)alloc(8192 * 4);              // encoder out (128x64)
  float* qbuf   = (float*)alloc(8192 * 4);
  int*   idxb   = (int*)  alloc(128 * 4);
  int*   cnts   = (int*)  alloc(512 * 4);
  float* lossA  = (float*)alloc(4);
  float* h0buf  = (float*)alloc(32768 * 4);             // (8*32, 128)
  float* xp1    = (float*)alloc(98304 * 4);             // (8*32, 384)
  float* hsF    = (float*)alloc(32768 * 4);             // (8,32,128)
  float* hsB    = (float*)alloc(32768 * 4);
  float* hcat1  = (float*)alloc(65536 * 4);             // (8*32, 256)
  float* hcat2  = (float*)alloc(65536 * 4);
  float* hp0    = (float*)alloc(16 * 896 * 4);          // state ping, 16 rows (8 live + 8 zero pad)
  float* hp1    = (float*)alloc(16 * 896 * 4);
  float* xcat   = (float*)alloc((size_t)16000 * 512 * 4);
  float* xp2    = (float*)alloc((size_t)16000 * 2688 * 4);
  float* hs2    = (float*)alloc((size_t)16000 * 896 * 4);
  float* f1     = (float*)alloc((size_t)16000 * 256 * 4);
  (void)ws_size; (void)n_in; (void)in_sizes; (void)out_size;

  auto fill = [&](float* p, long n, float v) {
    fill_f32<<<(unsigned)((n + 255) / 256), 256, 0, stream>>>(p, n, v);
  };
  auto gemm = [&](const float* A, const float* W, const float* bias, float* C,
                  int M, int N, int K, int lda, int ldw, int ldc, int act) {
    dim3 g((N + 63) / 64, (M + 31) / 32);
    gemm_bias_act<<<g, 256, 0, stream>>>(A, W, bias, C, M, N, K, lda, ldw, ldc, act);
  };

  // ================= Encoder: 5 x (im2col -> GEMM -> BN+ReLU), then 1x1 conv =================
  struct ConvCfg { int cin, k, stride, tin, tout; };
  const ConvCfg cfg[5] = {{80, 3, 1, 32, 32}, {512, 3, 1, 32, 32}, {512, 4, 2, 32, 16},
                          {512, 3, 1, 16, 16}, {512, 3, 1, 16, 16}};
  float* cur = nullptr;  // BTC activation buffer; layer0 reads mels (BCT) directly
  for (int L = 0; L < 5; ++L) {
    const ConvCfg& c = cfg[L];
    int M = 8 * c.tout, K = c.cin * c.k;
    long sb, sc, st;
    const float* src;
    if (L == 0) { src = mels; sb = (long)80 * 32; sc = 32; st = 1; }               // BCT
    else        { src = cur;  sb = (long)c.tin * 512; sc = 1; st = 512; }          // BTC (MxC)
    long tot = (long)8 * c.tout * c.cin * c.k;
    im2col_k<<<(unsigned)((tot + 255) / 256), 256, 0, stream>>>(
        src, colbuf, 8, c.cin, c.tin, c.tout, c.k, c.stride, 1, sb, sc, st);
    float* dst = (L & 1) ? actB : actA;
    gemm(colbuf, conv_ws[L], nullptr, dst, M, 512, K, K, K, 512, 0);
    bn_stats<<<512, 256, 0, stream>>>(dst, meanb, varb, M, 512);
    long mc = (long)M * 512;
    bn_relu_k<<<(unsigned)((mc + 255) / 256), 256, 0, stream>>>(dst, meanb, varb, bn_g[L], bn_b[L], M, 512);
    cur = dst;
  }
  // conv6 (1x1): z = cur(128x512) @ conv6_w(64x512)^T + b  -> rows (b*16+t), cols 64
  gemm(cur, conv6_w, conv6_b, zbuf, 128, 64, 512, 512, 512, 64, 0);

  // ================= VQ =================
  fill((float*)cnts, 512, 0.f);
  fill(lossA, 1, 0.f);
  vq_argmin<<<128, 128, 0, stream>>>(zbuf, codebook, idxb, cnts);
  vq_gather_loss<<<128, 64, 0, stream>>>(zbuf, codebook, idxb, qbuf, lossA);
  vq_finalize<<<1, 512, 0, stream>>>(cnts, lossA, out2);

  // ================= Vocoder =================
  // h0 = concat(repeat(q,2,axis=time), spk) -> (8*32, 128)
  build_h0<<<(32768 + 255) / 256, 256, 0, stream>>>(qbuf, spk_emb, speakers, h0buf);

  // Bi-GRU helper over S timesteps (input BTC layout, rows = b*S+t).
  // h state buffers hold 16 rows; rows 8..15 stay zero (unguarded WMMA A loads).
  auto run_gru = [&](const float* xin, int S, int I, int H,
                     const float* Wih, const float* Whh, const float* bih, const float* bhh,
                     float* xpbuf, float* hs, bool backward) {
    gemm(xin, Wih, bih, xpbuf, 8 * S, 3 * H, I, I, I, 3 * H, 0);  // xp = xin @ Wih^T + bih
    fill(hp0, 16L * H, 0.f);
    fill(hp1, 16L * H, 0.f);
    float* hp = hp0; float* hn = hp1;
    for (int s = 0; s < S; ++s) {
      int t = backward ? (S - 1 - s) : s;
      gru_step<<<H / 16, 32, 0, stream>>>(xpbuf + (long)t * 3 * H, (long)S * 3 * H,
                                          Whh, bhh, hp, hn,
                                          hs + (long)t * H, (long)S * H, H);
      float* tmp = hp; hp = hn; hn = tmp;
    }
  };

  // rnn1 layer 0: input (8,32,128) -> fwd/bwd H=128 -> concat (8,32,256)
  run_gru(h0buf, 32, 128, 128, r1[0][0], r1[0][1], r1[0][2], r1[0][3], xp1, hsF, false);
  run_gru(h0buf, 32, 128, 128, r1[1][0], r1[1][1], r1[1][2], r1[1][3], xp1, hsB, true);
  concat_h<<<(65536 + 255) / 256, 256, 0, stream>>>(hsF, hsB, hcat1, 8L * 32, 128);
  // rnn1 layer 1: input (8,32,256)
  run_gru(hcat1, 32, 256, 128, r1[2][0], r1[2][1], r1[2][2], r1[2][3], xp1, hsF, false);
  run_gru(hcat1, 32, 256, 128, r1[3][0], r1[3][1], r1[3][2], r1[3][3], xp1, hsB, true);
  concat_h<<<(65536 + 255) / 256, 256, 0, stream>>>(hsF, hsB, hcat2, 8L * 32, 128);

  // xcat = [aud_emb[x], repeat(hcat2[:,11:21,:], 200)] -> (16000, 512)
  {
    long tot = (long)16000 * 512;
    build_xcat<<<(unsigned)((tot + 255) / 256), 256, 0, stream>>>(x, aud_emb, hcat2, xcat);
  }

  // rnn2: I=512, H=896, S=2000 (xp precomputed as one big parallel GEMM)
  run_gru(xcat, 2000, 512, 896, Wih2, Whh2, bih2, bhh2, xp2, hs2, false);

  // fc1 (ReLU) and fc2 -> d_out
  gemm(hs2, fc1_w, fc1_b, f1, 16000, 256, 896, 896, 896, 256, 1);
  gemm(f1, fc2_w, fc2_b, out, 16000, 256, 256, 256, 256, 256, 0);
}
